// AttentionBlock_7507602833432
// MI455X (gfx1250) — matrix operand order repaired and checked by reading, not compiled, not run on hardware
//
#include <hip/hip_runtime.h>

// ---------------- problem constants ----------------
#define NB     4      // batch
#define CC     256    // channels
#define LL     4096   // H*W
#define NGROUP 32
#define CPG    8      // channels per group
#define NHEAD  4
#define DH     64     // head dim
#define EPSV   1e-5f

typedef __bf16 bf16;
typedef __attribute__((ext_vector_type(16))) __bf16 v16bf;
typedef __attribute__((ext_vector_type(8)))  __bf16 bf16x8;
typedef __attribute__((ext_vector_type(8)))  float  v8f;

// ---------------- WMMA helpers ----------------
__device__ __forceinline__ v8f wmma_bf16(v16bf a, v16bf b, v8f c) {
  // D = A(16x32 bf16) * B(32x16 bf16) + C(16x16 f32)
  return __builtin_amdgcn_wmma_f32_16x16x32_bf16(
      /*neg_a=*/false, a, /*neg_b=*/false, b,
      /*c_mod=*/(short)0, c, /*reuse_a=*/false, /*reuse_b=*/false);
}

// Load a 16-element-per-lane fragment as two 8x bf16 (16B) chunks at p and
// p+second_off elements. Works for global and LDS pointers (addrspace inferred).
__device__ __forceinline__ v16bf load_frag(const bf16* p, int second_off) {
  bf16x8 lo = *(const bf16x8*)(p);
  bf16x8 hi = *(const bf16x8*)(p + second_off);
  v16bf r;
#pragma unroll
  for (int i = 0; i < 8; ++i) { r[i] = lo[i]; r[i + 8] = hi[i]; }
  return r;
}

// half-wave (16-lane) reductions: rows of a C/D tile live in lanes 0-15 / 16-31
__device__ __forceinline__ float halfmax16(float v) {
  v = fmaxf(v, __shfl_xor(v, 1, 32));
  v = fmaxf(v, __shfl_xor(v, 2, 32));
  v = fmaxf(v, __shfl_xor(v, 4, 32));
  v = fmaxf(v, __shfl_xor(v, 8, 32));
  return v;
}
__device__ __forceinline__ float halfsum16(float v) {
  v += __shfl_xor(v, 1, 32);
  v += __shfl_xor(v, 2, 32);
  v += __shfl_xor(v, 4, 32);
  v += __shfl_xor(v, 8, 32);
  return v;
}

// ---------------- kernel 1: GroupNorm -> bf16, transposed [N][L][C] ----------------
__global__ __launch_bounds__(256) void gn_kernel(const float* __restrict__ x,
                                                 const float* __restrict__ gamma,
                                                 const float* __restrict__ beta,
                                                 bf16* __restrict__ xnT) {
  const int blk = blockIdx.x;            // n*NGROUP + g
  const int n = blk / NGROUP, g = blk % NGROUP;
  const int tid = threadIdx.x;
  const float* xb = x + ((size_t)n * CC + (size_t)g * CPG) * LL;

  float s = 0.f, ss = 0.f;
  for (int idx = tid; idx < CPG * LL; idx += 256) {   // coalesced
    float v = xb[idx];
    s += v; ss += v * v;
  }
  __shared__ float red0[256], red1[256];
  red0[tid] = s; red1[tid] = ss;
  __syncthreads();
  for (int off = 128; off > 0; off >>= 1) {
    if (tid < off) { red0[tid] += red0[tid + off]; red1[tid] += red1[tid + off]; }
    __syncthreads();
  }
  const float inv_n = 1.0f / (float)(CPG * LL);
  const float mean = red0[0] * inv_n;
  const float var  = red1[0] * inv_n - mean * mean;   // biased, matches jnp.var
  const float rstd = rsqrtf(var + EPSV);

  // pass 2: tid -> (c within group = tid&7, l stride 32); write xnT[n][l][c]
  const int ci = tid & 7;
  const int c  = g * CPG + ci;
  const float gm = gamma[c] * rstd;
  const float bt = beta[c] - mean * gm;
  const float* xr = x + ((size_t)n * CC + c) * LL;
  bf16* dst = xnT + (size_t)n * LL * CC + c;
  for (int l = (tid >> 3); l < LL; l += 32) {
    dst[(size_t)l * CC] = (bf16)(xr[l] * gm + bt);
  }
}

// ---------------- kernel 2: convert weights to bf16 ----------------
__global__ __launch_bounds__(256) void cvtw_kernel(const float* __restrict__ wq,
                                                   const float* __restrict__ wk,
                                                   const float* __restrict__ wv,
                                                   const float* __restrict__ wp,
                                                   bf16* __restrict__ q,
                                                   bf16* __restrict__ k,
                                                   bf16* __restrict__ v,
                                                   bf16* __restrict__ p) {
  int i = blockIdx.x * 256 + threadIdx.x;   // 65536 elements each
  q[i] = (bf16)wq[i];
  k[i] = (bf16)wk[i];
  v[i] = (bf16)wv[i];
  p[i] = (bf16)wp[i];
}

// ---------------- kernel 3: q/k/v projections via WMMA ----------------
// grid: x = L/256 (16), y = C/16 (16), z = n*3 + which (12); block = 128 (4 waves)
__global__ __launch_bounds__(128) void proj_kernel(const bf16* __restrict__ xnT,
                                                   const bf16* __restrict__ wq,
                                                   const bf16* __restrict__ wk,
                                                   const bf16* __restrict__ wv,
                                                   bf16* __restrict__ qb,
                                                   bf16* __restrict__ kb,
                                                   bf16* __restrict__ vT) {
  const int lane  = threadIdx.x & 31;
  const int wave  = threadIdx.x >> 5;
  const int which = blockIdx.z % 3;            // 0=q 1=k 2=v
  const int n     = blockIdx.z / 3;
  const int o0    = blockIdx.y * 16;
  const int l0    = blockIdx.x * 256 + wave * 64;

  const bf16* W = (which == 0) ? wq : ((which == 1) ? wk : wv);
  const bf16* B = xnT + (size_t)n * LL * CC;   // [L][C]

  const int m   = lane & 15;                   // A: M row
  const int akb = (lane >> 4) * 16;            // A: K base per lane-half
  const int nn  = lane & 15;                   // B: N col
  const int kb2 = (lane >> 4) * 16;            // B: K base per lane-half

  v8f acc[4] = {};
  for (int c0 = 0; c0 < CC; c0 += 32) {
    v16bf a = load_frag(W + (size_t)(o0 + m) * CC + c0 + akb, 8);
#pragma unroll
    for (int t = 0; t < 4; ++t) {
      v16bf b = load_frag(B + (size_t)(l0 + t * 16 + nn) * CC + c0 + kb2, 8);
      acc[t] = wmma_bf16(a, b, acc[t]);
    }
  }

  const int rr = (lane >> 4) * 8;              // D: row offset per lane-half
  const int h  = o0 / DH;
  const int d0 = o0 % DH;
#pragma unroll
  for (int t = 0; t < 4; ++t) {
    const int l = l0 + t * 16 + nn;
#pragma unroll
    for (int r = 0; r < 8; ++r) {
      const int d = d0 + r + rr;
      const bf16 val = (bf16)acc[t][r];
      if (which == 0)
        qb[(((size_t)n * NHEAD + h) * LL + l) * DH + d] = val;   // [n][h][l][d]
      else if (which == 1)
        kb[(((size_t)n * NHEAD + h) * LL + l) * DH + d] = val;   // [n][h][l][d]
      else
        vT[(((size_t)n * NHEAD + h) * DH + d) * LL + l] = val;   // [n][h][d][l]
    }
  }
}

// ---------------- kernel 4: flash attention ----------------
// grid: x = L/64 (64 row blocks), y = n*NHEAD + h (16); block = 128 (4 waves,
// each wave owns 16 query rows). Streams keys in tiles of 32 with online softmax.
__global__ __launch_bounds__(128) void attn_kernel(const bf16* __restrict__ qg,
                                                   const bf16* __restrict__ kg,
                                                   const bf16* __restrict__ vTg,
                                                   bf16* __restrict__ obuf) {
  const int nh   = blockIdx.y;
  const int n    = nh / NHEAD, h = nh % NHEAD;
  const int lane = threadIdx.x & 31;
  const int wave = threadIdx.x >> 5;
  const int i0   = blockIdx.x * 64 + wave * 16;

  const bf16* qb = qg  + (size_t)nh * LL * DH;   // [L][DH]
  const bf16* kb = kg  + (size_t)nh * LL * DH;   // [L][DH]
  const bf16* vb = vTg + (size_t)nh * DH * LL;   // [DH][L]

  __shared__ bf16 plds[4][16][32];               // per-wave P tile

  const int m   = lane & 15;
  const int akb = (lane >> 4) * 16;
  const int nn  = lane & 15;
  const int kb2 = (lane >> 4) * 16;
  const float scale = 0.125f;                    // dh^-0.5

  // Q A-fragments held in registers for the whole key sweep
  const v16bf qa0 = load_frag(qb + (size_t)(i0 + m) * DH + 0  + akb, 8);
  const v16bf qa1 = load_frag(qb + (size_t)(i0 + m) * DH + 32 + akb, 8);

  v8f oacc[4] = {};
  float mrow[8], lrow[8];
#pragma unroll
  for (int r = 0; r < 8; ++r) { mrow[r] = -1e30f; lrow[r] = 0.f; }

  for (int j0 = 0; j0 < LL; j0 += 32) {
    // ---- S = Q K^T for a 16x32 tile (two 16x16 D tiles) ----
    v8f s0 = {}, s1 = {};
    {
      v16bf b00 = load_frag(kb + (size_t)(j0 + 0  + nn) * DH + 0  + kb2, 8);
      v16bf b01 = load_frag(kb + (size_t)(j0 + 0  + nn) * DH + 32 + kb2, 8);
      s0 = wmma_bf16(qa0, b00, s0);
      s0 = wmma_bf16(qa1, b01, s0);
      v16bf b10 = load_frag(kb + (size_t)(j0 + 16 + nn) * DH + 0  + kb2, 8);
      v16bf b11 = load_frag(kb + (size_t)(j0 + 16 + nn) * DH + 32 + kb2, 8);
      s1 = wmma_bf16(qa0, b10, s1);
      s1 = wmma_bf16(qa1, b11, s1);
    }

    // ---- online softmax per row (row r lives in 16 lanes of one half-wave) ----
    float p0v[8], p1v[8];
#pragma unroll
    for (int r = 0; r < 8; ++r) {
      const float a0 = s0[r] * scale;
      const float a1 = s1[r] * scale;
      const float tmax = halfmax16(fmaxf(a0, a1));
      const float mnew = fmaxf(mrow[r], tmax);
      const float corr = __expf(mrow[r] - mnew);
      const float p0 = __expf(a0 - mnew);
      const float p1 = __expf(a1 - mnew);
      const float ps = halfsum16(p0 + p1);
      lrow[r] = lrow[r] * corr + ps;
      mrow[r] = mnew;
#pragma unroll
      for (int t = 0; t < 4; ++t) oacc[t][r] *= corr;
      p0v[r] = p0; p1v[r] = p1;
    }

    // ---- reshape P: D-layout -> A-layout via wave-private LDS tile ----
    const int rowoff = (lane >> 4) * 8;
#pragma unroll
    for (int r = 0; r < 8; ++r) {
      plds[wave][r + rowoff][nn]      = (bf16)p0v[r];
      plds[wave][r + rowoff][16 + nn] = (bf16)p1v[r];
    }
    __syncthreads();
    const v16bf pa = load_frag(&plds[wave][m][akb], 8);

    // ---- O += P * V  (B from vT: lane N = d fixed, K = j contiguous) ----
#pragma unroll
    for (int t = 0; t < 4; ++t) {
      v16bf bv = load_frag(vb + (size_t)(t * 16 + nn) * LL + j0 + kb2, 8);
      oacc[t] = wmma_bf16(pa, bv, oacc[t]);
    }
  }

  // ---- epilogue: normalize and store obuf[n][l][h*DH + d] (bf16) ----
  const int rowoff = (lane >> 4) * 8;
#pragma unroll
  for (int r = 0; r < 8; ++r) {
    const float inv = 1.0f / lrow[r];
    const int row = i0 + r + rowoff;
#pragma unroll
    for (int t = 0; t < 4; ++t) {
      const int d = t * 16 + nn;
      obuf[((size_t)n * LL + row) * CC + h * DH + d] = (bf16)(oacc[t][r] * inv);
    }
  }
}

// ---------------- kernel 5: output projection + bias (fp32 out) ----------------
// grid: x = L/256 (16), y = C/16 (16), z = n (4); block = 128 (4 waves)
__global__ __launch_bounds__(128) void outproj_kernel(const bf16* __restrict__ obuf,
                                                      const bf16* __restrict__ wp,
                                                      const float* __restrict__ bp,
                                                      float* __restrict__ out) {
  const int lane = threadIdx.x & 31;
  const int wave = threadIdx.x >> 5;
  const int n    = blockIdx.z;
  const int o0   = blockIdx.y * 16;
  const int l0   = blockIdx.x * 256 + wave * 64;

  const bf16* B = obuf + (size_t)n * LL * CC;  // [L][C]
  const int m   = lane & 15;
  const int akb = (lane >> 4) * 16;
  const int nn  = lane & 15;
  const int kb2 = (lane >> 4) * 16;

  v8f acc[4] = {};
  for (int c0 = 0; c0 < CC; c0 += 32) {
    v16bf a = load_frag(wp + (size_t)(o0 + m) * CC + c0 + akb, 8);
#pragma unroll
    for (int t = 0; t < 4; ++t) {
      v16bf b = load_frag(B + (size_t)(l0 + t * 16 + nn) * CC + c0 + kb2, 8);
      acc[t] = wmma_bf16(a, b, acc[t]);
    }
  }

  const int rr = (lane >> 4) * 8;
#pragma unroll
  for (int r = 0; r < 8; ++r) {
    const int o = o0 + r + rr;
    const float bias = bp[o];
#pragma unroll
    for (int t = 0; t < 4; ++t) {
      const int l = l0 + t * 16 + nn;
      out[((size_t)n * CC + o) * LL + l] = acc[t][r] + bias;   // [N][C][L]
    }
  }
}

// ---------------- host launcher ----------------
extern "C" void kernel_launch(void* const* d_in, const int* in_sizes, int n_in,
                              void* d_out, int out_size, void* d_ws, size_t ws_size,
                              hipStream_t stream) {
  const float* x     = (const float*)d_in[0];
  const float* gamma = (const float*)d_in[1];
  const float* beta  = (const float*)d_in[2];
  const float* Wq    = (const float*)d_in[3];
  const float* Wk    = (const float*)d_in[4];
  const float* Wv    = (const float*)d_in[5];
  const float* Wp    = (const float*)d_in[6];
  const float* bp    = (const float*)d_in[7];
  float* out = (float*)d_out;

  // workspace carve-up (bf16): 5 * N*L*C + 4 * C*C  ->  ~40.5 MB
  const size_t NLC = (size_t)NB * LL * CC;     // 4M elements
  bf16* xnT  = (bf16*)d_ws;
  bf16* qbuf = xnT  + NLC;
  bf16* kbuf = qbuf + NLC;
  bf16* vT   = kbuf + NLC;
  bf16* obuf = vT   + NLC;
  bf16* wqb  = obuf + NLC;
  bf16* wkb  = wqb  + (size_t)CC * CC;
  bf16* wvb  = wkb  + (size_t)CC * CC;
  bf16* wpb  = wvb  + (size_t)CC * CC;

  gn_kernel<<<NB * NGROUP, 256, 0, stream>>>(x, gamma, beta, xnT);
  cvtw_kernel<<<(CC * CC) / 256, 256, 0, stream>>>(Wq, Wk, Wv, Wp, wqb, wkb, wvb, wpb);
  proj_kernel<<<dim3(LL / 256, CC / 16, NB * 3), 128, 0, stream>>>(
      xnT, wqb, wkb, wvb, qbuf, kbuf, vT);
  attn_kernel<<<dim3(LL / 64, NB * NHEAD), 128, 0, stream>>>(qbuf, kbuf, vT, obuf);
  outproj_kernel<<<dim3(LL / 256, CC / 16, NB), 128, 0, stream>>>(obuf, wpb, bp, out);
}
